// CachedMistralLayer_62199716380854
// MI455X (gfx1250) — compile-verified, matrix operand order checked by reading, not run on hardware
//
#include <hip/hip_runtime.h>

typedef unsigned short u16;
typedef unsigned int   u32;

typedef __attribute__((ext_vector_type(16))) __bf16 v16bf;
typedef __attribute__((ext_vector_type(8)))  float  v8f;

#define DM    4096
#define KVD   1024
#define SEQ   1024
#define NB    2
#define NH    32
#define NKVH  8
#define HD    128
#define DMLP  14336
#define NTOK  (NB*SEQ)

// ---------- helpers ----------
__device__ __forceinline__ u16 f2bf(float f){
  u32 u = __float_as_uint(f);
  u32 r = u + 0x7FFFu + ((u >> 16) & 1u);   // round-to-nearest-even
  return (u16)(r >> 16);
}
__device__ __forceinline__ float bf2f(u16 h){
  return __uint_as_float(((u32)h) << 16);
}
// A/B fragment: elements 0..7 at p[0..7], elements 8..15 at p[16..23]
__device__ __forceinline__ v16bf load_frag(const u16* p){
  union { uint4 q[2]; v16bf v; } u;
  u.q[0] = *(const uint4*)(p);
  u.q[1] = *(const uint4*)(p + 16);
  return u.v;
}
__device__ __forceinline__ v8f wmma_bf16(v16bf a, v16bf b, v8f c){
  return __builtin_amdgcn_wmma_f32_16x16x32_bf16(false, a, false, b, (short)0, c, false, false);
}
__device__ __forceinline__ float d4(float4 a){ return a.x*a.x + a.y*a.y + a.z*a.z + a.w*a.w; }

// ---------- CDNA5 async global->LDS copy (ASYNCcnt-tracked, no VGPR staging) ----------
__device__ __forceinline__ void async_load_b128(void* lds_ptr, const void* gptr){
  u32 lds_off = (u32)(size_t)lds_ptr;       // low 32 bits of shared pointer = LDS offset
  asm volatile("global_load_async_to_lds_b128 %0, %1, off"
               :: "v"(lds_off), "v"(gptr) : "memory");
}
__device__ __forceinline__ void wait_async0(){
  asm volatile("s_wait_asynccnt 0x0" ::: "memory");
}

// ---------- RMSNorm: fp32 row -> bf16 row ----------
__global__ __launch_bounds__(256) void rmsnorm_bf16_kernel(
    const float* __restrict__ x, const float* __restrict__ w, u16* __restrict__ y)
{
  int tid = threadIdx.x;
  size_t rowoff = (size_t)blockIdx.x * DM;
  const float4* xr = (const float4*)(x + rowoff + tid*16);
  float4 v0 = xr[0], v1 = xr[1], v2 = xr[2], v3 = xr[3];
  float ss = d4(v0) + d4(v1) + d4(v2) + d4(v3);
  for (int m = 16; m >= 1; m >>= 1) ss += __shfl_xor(ss, m, 32);
  __shared__ float red[9];
  if ((tid & 31) == 0) red[tid >> 5] = ss;
  __syncthreads();
  if (tid == 0){ float t = 0.f; for (int j = 0; j < 8; ++j) t += red[j]; red[8] = t; }
  __syncthreads();
  float rms = rsqrtf(red[8] * (1.0f / DM) + 1e-5f);
  const float4* wr = (const float4*)(w + tid*16);
  float xv[16], wv[16];
  ((float4*)xv)[0]=v0; ((float4*)xv)[1]=v1; ((float4*)xv)[2]=v2; ((float4*)xv)[3]=v3;
  ((float4*)wv)[0]=wr[0]; ((float4*)wv)[1]=wr[1]; ((float4*)wv)[2]=wr[2]; ((float4*)wv)[3]=wr[3];
  union { u16 s[16]; uint4 q[2]; } o;
  #pragma unroll
  for (int j = 0; j < 16; ++j) o.s[j] = f2bf(xv[j] * rms * wv[j]);
  uint4* yq = (uint4*)(y + rowoff + tid*16);
  yq[0] = o.q[0]; yq[1] = o.q[1];
}

// ---------- weight convert+transpose: f32 [K][N] -> bf16 [N][K] ----------
__global__ __launch_bounds__(256) void wtrans_bf16_kernel(
    const float* __restrict__ src, u16* __restrict__ dst, int K, int N)
{
  __shared__ float tile[32][33];
  int tx = threadIdx.x, ty = threadIdx.y;
  size_t n0 = (size_t)blockIdx.x * 32, k0 = (size_t)blockIdx.y * 32;
  #pragma unroll
  for (int r = ty; r < 32; r += 8) tile[r][tx] = src[(k0 + r) * (size_t)N + n0 + tx];
  __syncthreads();
  #pragma unroll
  for (int r = ty; r < 32; r += 8) dst[(n0 + r) * (size_t)K + k0 + tx] = f2bf(tile[tx][r]);
}

// ---------- RoPE (+ optional attn scale): f32 [T][nh][128] -> bf16 ----------
__global__ __launch_bounds__(256) void rope_bf16_kernel(
    const float* __restrict__ src, u16* __restrict__ dst, int nheads, float outscale)
{
  int idx = blockIdx.x * 256 + threadIdx.x;
  int total = NTOK * nheads * 64;
  if (idx >= total) return;
  int i = idx & 63;
  int h = (idx >> 6) % nheads;
  int t = idx / (64 * nheads);
  int s = t & (SEQ - 1);                       // position in sequence
  size_t base = ((size_t)t * nheads + h) * HD;
  float x1 = src[base + i], x2 = src[base + 64 + i];
  float inv = __expf(-0.14391157f * (float)i); // ln(10000)/64
  float ang = (float)s * inv;
  float sn, cs; __sincosf(ang, &sn, &cs);
  dst[base + i]      = f2bf((x1 * cs - x2 * sn) * outscale);
  dst[base + 64 + i] = f2bf((x2 * cs + x1 * sn) * outscale);
}

// ---------- V convert+transpose: f32 [b][s][kvh][hd] -> bf16 [b][kvh][hd][s] ----------
__global__ __launch_bounds__(256) void vtrans_bf16_kernel(
    const float* __restrict__ vf, u16* __restrict__ vbT)
{
  int idx = blockIdx.x * 256 + threadIdx.x;   // flat index into vbT
  int s  = idx & (SEQ - 1);
  int hd = (idx >> 10) & (HD - 1);
  int kh = (idx >> 17) & 7;
  int b  = idx >> 20;
  vbT[idx] = f2bf(vf[((size_t)(b * SEQ + s) * NKVH + kh) * HD + hd]);
}

// ---------- bf16 WMMA GEMM: C[M,N] = A[M,K] * Bt[N,K]^T ----------
// Double-buffered LDS, async global->LDS staging, fused epilogues.
// mode 0: Cf = acc             mode 1: Cf = acc + resid
// mode 2: Cb = bf16(silu(acc)) mode 3: Cb = bf16(bf2f(gate) * acc)
__global__ __launch_bounds__(256) void gemm_bf16_kernel(
    const u16* __restrict__ A, const u16* __restrict__ Bt,
    float* __restrict__ Cf, u16* __restrict__ Cb,
    const float* __restrict__ resid, const u16* __restrict__ gate,
    int M, int N, int K, int mode)
{
  __shared__ __align__(16) u16 As[2][128 * 40];
  __shared__ __align__(16) u16 Bs[2][128 * 40];
  int tid = threadIdx.x, lane = tid & 31, wid = tid >> 5;
  int ln15 = lane & 15, h8 = lane >> 4;
  int wm = wid & 3, wn = wid >> 2;
  size_t m0 = (size_t)blockIdx.y * 128, n0 = (size_t)blockIdx.x * 128;

  // this thread's two staging slots (row, k-segment of 8 bf16)
  int idx0 = tid,       row0 = idx0 >> 2, seg0 = idx0 & 3;
  int idx1 = tid + 256, row1 = idx1 >> 2, seg1 = idx1 & 3;

  auto stage = [&](int kt, int buf){
    int k0 = kt << 5;
    async_load_b128(&As[buf][row0 * 40 + seg0 * 8], &A [(m0 + row0) * (size_t)K + k0 + seg0 * 8]);
    async_load_b128(&Bs[buf][row0 * 40 + seg0 * 8], &Bt[(n0 + row0) * (size_t)K + k0 + seg0 * 8]);
    async_load_b128(&As[buf][row1 * 40 + seg1 * 8], &A [(m0 + row1) * (size_t)K + k0 + seg1 * 8]);
    async_load_b128(&Bs[buf][row1 * 40 + seg1 * 8], &Bt[(n0 + row1) * (size_t)K + k0 + seg1 * 8]);
  };

  v8f zv = {0.f,0.f,0.f,0.f,0.f,0.f,0.f,0.f};
  v8f acc[2][4];
  #pragma unroll
  for (int mt = 0; mt < 2; ++mt)
    #pragma unroll
    for (int nt = 0; nt < 4; ++nt) acc[mt][nt] = zv;

  int nk = K >> 5;
  stage(0, 0);
  for (int kt = 0; kt < nk; ++kt){
    int buf = kt & 1;
    wait_async0();          // own async writes for `buf` have landed
    __syncthreads();        // everyone's have landed; previous readers of buf^1 are done
    if (kt + 1 < nk) stage(kt + 1, buf ^ 1);   // prefetch next tile into other buffer

    v16bf af[2], bfr[4];
    #pragma unroll
    for (int mt = 0; mt < 2; ++mt) af[mt]  = load_frag(&As[buf][(wm * 32 + mt * 16 + ln15) * 40 + h8 * 8]);
    #pragma unroll
    for (int nt = 0; nt < 4; ++nt) bfr[nt] = load_frag(&Bs[buf][(wn * 64 + nt * 16 + ln15) * 40 + h8 * 8]);
    #pragma unroll
    for (int mt = 0; mt < 2; ++mt)
      #pragma unroll
      for (int nt = 0; nt < 4; ++nt)
        acc[mt][nt] = wmma_bf16(af[mt], bfr[nt], acc[mt][nt]);
    __syncthreads();        // all waves done reading `buf` before it is refilled
  }

  #pragma unroll
  for (int mt = 0; mt < 2; ++mt)
    #pragma unroll
    for (int nt = 0; nt < 4; ++nt)
      #pragma unroll
      for (int i = 0; i < 8; ++i){
        size_t m = m0 + wm * 32 + mt * 16 + h8 * 8 + i;
        size_t n = n0 + wn * 64 + nt * 16 + ln15;
        size_t gi = m * (size_t)N + n;
        float v = acc[mt][nt][i];
        if (mode == 0)      Cf[gi] = v;
        else if (mode == 1) Cf[gi] = v + resid[gi];
        else if (mode == 2) Cb[gi] = f2bf(v * __builtin_amdgcn_rcpf(1.0f + __expf(-v)));
        else                Cb[gi] = f2bf(bf2f(gate[gi]) * v);
      }
}

// ---------- Flash attention (causal, GQA 4:1), all-WMMA ----------
// qb: bf16 [T][32][128] (RoPE + 1/sqrt(128) pre-applied)
// kb: bf16 [T][8][128]  (RoPE applied)
// vbT: bf16 [b][kvh][hd][S]
// ob: bf16 [T][32][128]
__global__ __launch_bounds__(128) void flash_attn_kernel(
    const u16* __restrict__ qb, const u16* __restrict__ kb,
    const u16* __restrict__ vbT, u16* __restrict__ ob)
{
  __shared__ __align__(16) u16 Ks[64 * 136];   // [key][hd]
  __shared__ __align__(16) u16 Vs[128 * 72];   // [hd][key]
  __shared__ __align__(16) u16 Ps[64 * 72];    // [qrow][key]
  int tid = threadIdx.x, lane = tid & 31, wid = tid >> 5;
  int ln15 = lane & 15, h8 = lane >> 4;
  int qblk = blockIdx.x;
  int bh = blockIdx.y;
  int b = bh >> 5;
  int h = bh & 31;
  int kvh = h >> 2;

  // preload Q fragments (A-matrix: lane = row)
  size_t qrow = (size_t)(b * SEQ + qblk * 64 + wid * 16 + ln15);
  const u16* qp = qb + qrow * DM + h * HD;
  v16bf qfrag[4];
  #pragma unroll
  for (int kk = 0; kk < 4; ++kk) qfrag[kk] = load_frag(qp + kk * 32 + h8 * 8);

  v8f zv = {0.f,0.f,0.f,0.f,0.f,0.f,0.f,0.f};
  v8f oacc[8];
  #pragma unroll
  for (int nto = 0; nto < 8; ++nto) oacc[nto] = zv;
  float rowmax[8], rowsum[8];
  #pragma unroll
  for (int i = 0; i < 8; ++i){ rowmax[i] = -3.0e38f; rowsum[i] = 0.f; }

  for (int kbk = 0; kbk <= qblk; ++kbk){
    __syncthreads();   // previous iteration's readers of Ks/Vs/Ps are done
    // async-stage K block (row-major, hd contiguous)
    size_t kbase = ((size_t)(b * SEQ + kbk * 64)) * (NKVH * HD) + kvh * HD;
    #pragma unroll
    for (int it = 0; it < 8; ++it){
      int idx = tid + it * 128;
      int row = idx >> 4, seg = idx & 15;
      async_load_b128(&Ks[row * 136 + seg * 8], &kb[kbase + (size_t)row * (NKVH * HD) + seg * 8]);
    }
    // async-stage V block (hd-major: key contiguous, from pre-transposed vbT)
    size_t vbase = ((size_t)(b * NKVH + kvh) * HD) * SEQ + kbk * 64;
    #pragma unroll
    for (int it = 0; it < 8; ++it){
      int idx = tid + it * 128;
      int row = idx >> 3, seg = idx & 7;
      async_load_b128(&Vs[row * 72 + seg * 8], &vbT[vbase + (size_t)row * SEQ + seg * 8]);
    }
    wait_async0();
    __syncthreads();

    // S = Q * K^T  (16 q rows x 64 keys per wave)
    v8f sacc[4];
    #pragma unroll
    for (int nt = 0; nt < 4; ++nt) sacc[nt] = zv;
    #pragma unroll
    for (int nt = 0; nt < 4; ++nt)
      #pragma unroll
      for (int kk = 0; kk < 4; ++kk){
        v16bf kf = load_frag(&Ks[(nt * 16 + ln15) * 136 + kk * 32 + h8 * 8]);
        sacc[nt] = wmma_bf16(qfrag[kk], kf, sacc[nt]);
      }

    if (kbk == qblk){
      #pragma unroll
      for (int nt = 0; nt < 4; ++nt)
        #pragma unroll
        for (int i = 0; i < 8; ++i){
          int key = nt * 16 + ln15;
          int qr  = wid * 16 + h8 * 8 + i;
          if (key > qr) sacc[nt][i] = -1.0e30f;
        }
    }

    // online softmax (row stats reduced across the 16 lanes of each half)
    #pragma unroll
    for (int i = 0; i < 8; ++i){
      float tm = fmaxf(fmaxf(sacc[0][i], sacc[1][i]), fmaxf(sacc[2][i], sacc[3][i]));
      tm = fmaxf(tm, __shfl_xor(tm, 1, 32));
      tm = fmaxf(tm, __shfl_xor(tm, 2, 32));
      tm = fmaxf(tm, __shfl_xor(tm, 4, 32));
      tm = fmaxf(tm, __shfl_xor(tm, 8, 32));
      float nm = fmaxf(rowmax[i], tm);
      float sc = __expf(rowmax[i] - nm);
      rowmax[i] = nm;
      float ps = 0.f;
      #pragma unroll
      for (int nt = 0; nt < 4; ++nt){
        float e = __expf(sacc[nt][i] - nm);
        sacc[nt][i] = e;
        ps += e;
      }
      ps += __shfl_xor(ps, 1, 32);
      ps += __shfl_xor(ps, 2, 32);
      ps += __shfl_xor(ps, 4, 32);
      ps += __shfl_xor(ps, 8, 32);
      rowsum[i] = rowsum[i] * sc + ps;
      #pragma unroll
      for (int nto = 0; nto < 8; ++nto) oacc[nto][i] *= sc;
    }

    // transpose P (acc layout -> A-fragment layout) through LDS
    #pragma unroll
    for (int nt = 0; nt < 4; ++nt)
      #pragma unroll
      for (int i = 0; i < 8; ++i)
        Ps[(wid * 16 + h8 * 8 + i) * 72 + nt * 16 + ln15] = f2bf(sacc[nt][i]);
    __syncthreads();

    // O += P * V
    #pragma unroll
    for (int kk2 = 0; kk2 < 2; ++kk2){
      v16bf pf = load_frag(&Ps[(wid * 16 + ln15) * 72 + kk2 * 32 + h8 * 8]);
      #pragma unroll
      for (int nto = 0; nto < 8; ++nto){
        v16bf vfr = load_frag(&Vs[(nto * 16 + ln15) * 72 + kk2 * 32 + h8 * 8]);
        oacc[nto] = wmma_bf16(pf, vfr, oacc[nto]);
      }
    }
  }

  // normalize + store bf16 O
  #pragma unroll
  for (int i = 0; i < 8; ++i){
    float inv = __builtin_amdgcn_rcpf(rowsum[i]);
    size_t tok = (size_t)(b * SEQ + qblk * 64 + wid * 16 + h8 * 8 + i);
    #pragma unroll
    for (int nto = 0; nto < 8; ++nto)
      ob[tok * DM + h * HD + nto * 16 + ln15] = f2bf(oacc[nto][i] * inv);
  }
}

// ---------- host ----------
extern "C" void kernel_launch(void* const* d_in, const int* in_sizes, int n_in,
                              void* d_out, int out_size, void* d_ws, size_t ws_size,
                              hipStream_t stream)
{
  (void)in_sizes; (void)n_in; (void)out_size; (void)ws_size;
  const float* stm  = (const float*)d_in[0];
  const float* Wq   = (const float*)d_in[1];
  const float* Wk   = (const float*)d_in[2];
  const float* Wv   = (const float*)d_in[3];
  const float* Wo   = (const float*)d_in[4];
  const float* Wg   = (const float*)d_in[5];
  const float* Wu   = (const float*)d_in[6];
  const float* Wd   = (const float*)d_in[7];
  const float* wln1 = (const float*)d_in[8];
  const float* wln2 = (const float*)d_in[9];
  float* out = (float*)d_out;

  char* ws = (char*)d_ws;
  size_t off = 0;
  auto alloc = [&](size_t bytes) -> void* {
    void* p = ws + off;
    off += (bytes + 255) & ~(size_t)255;
    return p;
  };
  u16*   xln  = (u16*)  alloc((size_t)NTOK * DM   * 2);
  u16*   WqT  = (u16*)  alloc((size_t)DM   * DM   * 2);
  u16*   WkT  = (u16*)  alloc((size_t)DM   * KVD  * 2);
  u16*   WvT  = (u16*)  alloc((size_t)DM   * KVD  * 2);
  u16*   WoT  = (u16*)  alloc((size_t)DM   * DM   * 2);
  u16*   WgT  = (u16*)  alloc((size_t)DM   * DMLP * 2);
  u16*   WuT  = (u16*)  alloc((size_t)DM   * DMLP * 2);
  u16*   WdT  = (u16*)  alloc((size_t)DMLP * DM   * 2);
  float* qf   = (float*)alloc((size_t)NTOK * DM   * 4);
  float* kf   = (float*)alloc((size_t)NTOK * KVD  * 4);
  float* vf   = (float*)alloc((size_t)NTOK * KVD  * 4);
  u16*   qbb  = (u16*)  alloc((size_t)NTOK * DM   * 2);
  u16*   kbb  = (u16*)  alloc((size_t)NTOK * KVD  * 2);
  u16*   vbT  = (u16*)  alloc((size_t)NTOK * KVD  * 2);
  u16*   obb  = (u16*)  alloc((size_t)NTOK * DM   * 2);
  float* stm1 = (float*)alloc((size_t)NTOK * DM   * 4);
  u16*   x2   = (u16*)  alloc((size_t)NTOK * DM   * 2);
  u16*   gb   = (u16*)  alloc((size_t)NTOK * DMLP * 2);
  u16*   hb   = (u16*)  alloc((size_t)NTOK * DMLP * 2);

  dim3 tb(32, 8);
  wtrans_bf16_kernel<<<dim3(DM/32,   DM/32),   tb, 0, stream>>>(Wq, WqT, DM,   DM);
  wtrans_bf16_kernel<<<dim3(KVD/32,  DM/32),   tb, 0, stream>>>(Wk, WkT, DM,   KVD);
  wtrans_bf16_kernel<<<dim3(KVD/32,  DM/32),   tb, 0, stream>>>(Wv, WvT, DM,   KVD);
  wtrans_bf16_kernel<<<dim3(DM/32,   DM/32),   tb, 0, stream>>>(Wo, WoT, DM,   DM);
  wtrans_bf16_kernel<<<dim3(DMLP/32, DM/32),   tb, 0, stream>>>(Wg, WgT, DM,   DMLP);
  wtrans_bf16_kernel<<<dim3(DMLP/32, DM/32),   tb, 0, stream>>>(Wu, WuT, DM,   DMLP);
  wtrans_bf16_kernel<<<dim3(DM/32,   DMLP/32), tb, 0, stream>>>(Wd, WdT, DMLP, DM);

  rmsnorm_bf16_kernel<<<NTOK, 256, 0, stream>>>(stm, wln1, xln);

  gemm_bf16_kernel<<<dim3(DM/128,  NTOK/128), 256, 0, stream>>>(xln, WqT, qf, nullptr, nullptr, nullptr, NTOK, DM,  DM, 0);
  gemm_bf16_kernel<<<dim3(KVD/128, NTOK/128), 256, 0, stream>>>(xln, WkT, kf, nullptr, nullptr, nullptr, NTOK, KVD, DM, 0);
  gemm_bf16_kernel<<<dim3(KVD/128, NTOK/128), 256, 0, stream>>>(xln, WvT, vf, nullptr, nullptr, nullptr, NTOK, KVD, DM, 0);

  rope_bf16_kernel<<<(NTOK*NH*64)/256,   256, 0, stream>>>(qf, qbb, NH,   0.08838834764831845f);
  rope_bf16_kernel<<<(NTOK*NKVH*64)/256, 256, 0, stream>>>(kf, kbb, NKVH, 1.0f);
  vtrans_bf16_kernel<<<(NB*NKVH*HD*SEQ)/256, 256, 0, stream>>>(vf, vbT);

  flash_attn_kernel<<<dim3(SEQ/64, NB*NH), 128, 0, stream>>>(qbb, kbb, vbT, obb);

  gemm_bf16_kernel<<<dim3(DM/128, NTOK/128), 256, 0, stream>>>(obb, WoT, stm1, nullptr, stm, nullptr, NTOK, DM, DM, 1);

  rmsnorm_bf16_kernel<<<NTOK, 256, 0, stream>>>(stm1, wln2, x2);

  gemm_bf16_kernel<<<dim3(DMLP/128, NTOK/128), 256, 0, stream>>>(x2, WgT, nullptr, gb, nullptr, nullptr, NTOK, DMLP, DM, 2);
  gemm_bf16_kernel<<<dim3(DMLP/128, NTOK/128), 256, 0, stream>>>(x2, WuT, nullptr, hb, nullptr, gb,      NTOK, DMLP, DM, 3);
  gemm_bf16_kernel<<<dim3(DM/128,   NTOK/128), 256, 0, stream>>>(hb, WdT, out, nullptr, stm1, nullptr,   NTOK, DM, DMLP, 1);
}
